// LSTM_44624710205763
// MI455X (gfx1250) — matrix operand order repaired and checked by reading, not compiled, not run on hardware
//
#include <hip/hip_runtime.h>
#include <hip/hip_bf16.h>
#include <cstdint>

// LSTM: B=128, S=512, I=300, H=512, gates=4H=2048.
// bf16 WMMA (V_WMMA_F32_16X16X32_BF16) with f32 accumulate.
// CDNA5 data path: global_load_async_to_lds_b128 staging + ds_load_b128 frags.

#define B_   128
#define S_   512
#define I_   300
#define IP_  320      // I padded to multiple of 32 (WMMA K step)
#define H_   512
#define G4_  2048

typedef __attribute__((ext_vector_type(16))) __bf16       v16bf;
typedef __attribute__((ext_vector_type(8)))  float        v8f;
typedef __attribute__((ext_vector_type(4)))  unsigned int v4u;

union FragBF {
    v16bf v;
    v4u   q[2];
};

__device__ __forceinline__ FragBF load_frag(const __bf16* p) {
    FragBF f;
    f.q[0] = *(const v4u*)(p);
    f.q[1] = *(const v4u*)(p + 16);
    return f;
}

__device__ __forceinline__ float sigf(float x) {
    return 1.0f / (1.0f + __expf(-x));
}

// Async copy of one 16-byte chunk into LDS (per-lane). Tracked by ASYNCcnt.
__device__ __forceinline__ void async_cp16(unsigned ldsoff, unsigned long long ga) {
    asm volatile("global_load_async_to_lds_b128 %0, %1, off"
                 :: "v"(ldsoff), "v"(ga) : "memory");
}
__device__ __forceinline__ void wait_async0() {
    asm volatile("s_wait_asynccnt 0" ::: "memory");
}

// ---------------------------------------------------------------------------
// Conversion kernels (bf16 operand build; weights transposed so B-fragment
// loads are contiguous 8-half runs along K).
// ---------------------------------------------------------------------------
__global__ __launch_bounds__(256) void cvt_x_bf16(const float* __restrict__ x,
                                                  __bf16* __restrict__ xb) {
    const size_t idx = (size_t)blockIdx.x * blockDim.x + threadIdx.x;
    if (idx >= (size_t)B_ * S_ * IP_) return;
    const size_t m = idx / IP_;
    const int    k = (int)(idx % IP_);
    xb[idx] = (k < I_) ? (__bf16)x[m * I_ + k] : (__bf16)0.0f;
}

__global__ __launch_bounds__(256) void cvt_w_bf16(const float* __restrict__ W,
                                                  __bf16* __restrict__ Wt) {
    const size_t idx = (size_t)blockIdx.x * blockDim.x + threadIdx.x;
    if (idx >= (size_t)G4_ * IP_) return;
    const size_t g = idx / IP_;
    const int    k = (int)(idx % IP_);
    Wt[idx] = (k < I_) ? (__bf16)W[(size_t)k * G4_ + g] : (__bf16)0.0f;
}

__global__ __launch_bounds__(256) void cvt_u_bf16(const float* __restrict__ U,
                                                  __bf16* __restrict__ Ut) {
    const size_t idx = (size_t)blockIdx.x * blockDim.x + threadIdx.x;
    if (idx >= (size_t)G4_ * H_) return;
    const size_t g = idx / H_;
    const int    k = (int)(idx % H_);
    Ut[idx] = (__bf16)U[(size_t)k * G4_ + g];
}

__global__ __launch_bounds__(256) void init_state(__bf16* __restrict__ h0,
                                                  __bf16* __restrict__ h1,
                                                  float* __restrict__ c) {
    const int idx = blockIdx.x * blockDim.x + threadIdx.x;
    if (idx >= B_ * H_) return;
    h0[idx] = (__bf16)0.0f;
    h1[idx] = (__bf16)0.0f;
    c[idx]  = 0.0f;
}

// ---------------------------------------------------------------------------
// GEMM 1: xw[s][b][g] = bias[g] + sum_k x[b,s,k] * W[k,g]
// WG = 8 waves: tile 32(M) x 512(N); wave tile 32 x 64 (2 M sub-tiles, so
// every B fragment feeds two WMMAs). The 32x320 A panel (20,480 contiguous
// bytes) is async-staged into LDS once; K loop is barrier-free.
// ---------------------------------------------------------------------------
__global__ __launch_bounds__(256) void lstm_xw_gemm(const __bf16* __restrict__ xb,
                                                    const __bf16* __restrict__ Wt,
                                                    const float*  __restrict__ bias,
                                                    float*        __restrict__ xw) {
    __shared__ __bf16 sA[32][IP_];   // 20,480 B

    const int tid  = threadIdx.x;
    const int wave = tid >> 5;
    const int lane = tid & 31;
    const int lo   = lane & 15;
    const int hi   = lane >> 4;
    const int m0   = blockIdx.x * 32;
    const int n0   = blockIdx.y * 512 + wave * 64;

    // Async-stage the full 32x320 A panel: contiguous 1280 x 16B chunks.
    {
        const unsigned long long src = (unsigned long long)(xb + (size_t)m0 * IP_);
        const unsigned           lds = (unsigned)(uintptr_t)&sA[0][0];
        for (int d = tid; d < (32 * IP_ * 2) / 16; d += 256) {
            async_cp16(lds + d * 16, src + (unsigned long long)d * 16);
        }
        wait_async0();
    }
    __syncthreads();

    v8f acc[2][4] = {};

    const __bf16* wbase = Wt + (size_t)(n0 + lo) * IP_ + hi * 8;

#pragma unroll
    for (int kk = 0; kk < IP_ / 32; ++kk) {
        FragBF a0, a1;
        a0.q[0] = *(const v4u*)&sA[lo][kk * 32 + hi * 8];
        a0.q[1] = *(const v4u*)&sA[lo][kk * 32 + hi * 8 + 16];
        a1.q[0] = *(const v4u*)&sA[16 + lo][kk * 32 + hi * 8];
        a1.q[1] = *(const v4u*)&sA[16 + lo][kk * 32 + hi * 8 + 16];
#pragma unroll
        for (int g = 0; g < 4; ++g) {
            const FragBF bfr = load_frag(wbase + (size_t)g * 16 * IP_ + kk * 32);
            acc[0][g] = __builtin_amdgcn_wmma_f32_16x16x32_bf16(
                false, a0.v, false, bfr.v, (short)0, acc[0][g], false, false);
            acc[1][g] = __builtin_amdgcn_wmma_f32_16x16x32_bf16(
                false, a1.v, false, bfr.v, (short)0, acc[1][g], false, false);
        }
    }

    // Epilogue: + bias, write xw in (s, b, g) order for the scan.
    const int bb = m0 >> 9;          // m0 / S_ (tile stays within one batch row)
    const int s0 = m0 & (S_ - 1);
#pragma unroll
    for (int mt = 0; mt < 2; ++mt) {
#pragma unroll
        for (int c = 0; c < 4; ++c) {
            const int   g  = n0 + c * 16 + lo;
            const float bv = bias[g];
#pragma unroll
            for (int v = 0; v < 8; ++v) {
                const int s = s0 + mt * 16 + v + 8 * hi;   // D: M = v + 8*(lane>=16)
                xw[((size_t)s * B_ + bb) * G4_ + g] = acc[mt][c][v] + bv;
            }
        }
    }
}

// ---------------------------------------------------------------------------
// Recurrent step: gates = xw[s] + h @ U + fused pointwise update.
// WG = 4 waves; wave tile 16(batch) x 16(h-col) x 4 gates. The 64 KB Ut
// j-panel (shared by all 4 waves) is async-staged into LDS once per step;
// A (h) fragments stream from L2 double-buffered; c preloaded / xw
// prefetched during the staging window.
// ---------------------------------------------------------------------------
__global__ __launch_bounds__(128) void lstm_step(const float*  __restrict__ xw,
                                                 const __bf16* __restrict__ Ut,
                                                 const __bf16* __restrict__ h_in,
                                                 __bf16*       __restrict__ h_out,
                                                 float*        __restrict__ cst,
                                                 float*        __restrict__ out,
                                                 int s) {
    __shared__ __bf16 sU[4][16][H_];   // 64 KB: 4 gates x 16 j-cols x K=512

    const int tid  = threadIdx.x;
    const int wave = tid >> 5;
    const int lane = tid & 31;
    const int lo   = lane & 15;
    const int hi   = lane >> 4;
    const int m0   = blockIdx.x * 64 + wave * 16;
    const int j0   = blockIdx.y * 16;
    const int j    = j0 + lo;

    {
        const unsigned lds = (unsigned)(uintptr_t)&sU[0][0][0];
#pragma unroll
        for (int gt = 0; gt < 4; ++gt) {
            const unsigned long long src =
                (unsigned long long)(Ut + (size_t)(gt * H_ + j0) * H_);
            const unsigned base = lds + gt * 16 * H_ * 2;
            for (int d = tid; d < (16 * H_ * 2) / 16; d += 128) {
                async_cp16(base + d * 16, src + (unsigned long long)d * 16);
            }
        }
    }

    // Overlap with staging: preload c, prefetch xw gate rows.
    float cv[8];
#pragma unroll
    for (int v = 0; v < 8; ++v) {
        const int m = m0 + v + 8 * hi;
        cv[v] = cst[m * H_ + j];
        const size_t xr = ((size_t)s * B_ + m) * G4_;
        __builtin_prefetch(&xw[xr + j], 0, 0);
        __builtin_prefetch(&xw[xr + H_ + j], 0, 0);
        __builtin_prefetch(&xw[xr + 2 * H_ + j], 0, 0);
        __builtin_prefetch(&xw[xr + 3 * H_ + j], 0, 0);
    }

    wait_async0();
    __syncthreads();

    v8f    acc[4] = {};
    FragBF a[2];

    const __bf16* hrow = h_in + (size_t)(m0 + lo) * H_ + hi * 8;
    a[0].q[0] = *(const v4u*)(hrow);
    a[0].q[1] = *(const v4u*)(hrow + 16);

#pragma unroll
    for (int kk = 0; kk < H_ / 32; ++kk) {
        const int cur = kk & 1;
        if (kk + 1 < H_ / 32) {
            a[cur ^ 1].q[0] = *(const v4u*)(hrow + (kk + 1) * 32);
            a[cur ^ 1].q[1] = *(const v4u*)(hrow + (kk + 1) * 32 + 16);
        }
        FragBF bf[4];
#pragma unroll
        for (int gt = 0; gt < 4; ++gt) {
            bf[gt].q[0] = *(const v4u*)&sU[gt][lo][kk * 32 + hi * 8];
            bf[gt].q[1] = *(const v4u*)&sU[gt][lo][kk * 32 + hi * 8 + 16];
        }
#pragma unroll
        for (int gt = 0; gt < 4; ++gt)
            acc[gt] = __builtin_amdgcn_wmma_f32_16x16x32_bf16(
                false, a[cur].v, false, bf[gt].v, (short)0, acc[gt], false, false);
    }

#pragma unroll
    for (int v = 0; v < 8; ++v) {
        const int    m  = m0 + v + 8 * hi;
        const size_t xr = ((size_t)s * B_ + m) * G4_;
        const float gi = acc[0][v] + xw[xr + j];
        const float gf = acc[1][v] + xw[xr + H_ + j];
        const float gg = acc[2][v] + xw[xr + 2 * H_ + j];
        const float go = acc[3][v] + xw[xr + 3 * H_ + j];

        const float iv = sigf(gi);
        const float fv = sigf(gf);
        const float gv = tanhf(gg);
        const float ov = sigf(go);

        const float cn = fv * cv[v] + iv * gv;
        const int   ci = m * H_ + j;
        cst[ci] = cn;
        const float hv = ov * tanhf(cn);
        h_out[ci] = (__bf16)hv;
        out[((size_t)m * S_ + s) * H_ + j] = hv;
    }
}

// ---------------------------------------------------------------------------
// Host launch
// ---------------------------------------------------------------------------
extern "C" void kernel_launch(void* const* d_in, const int* in_sizes, int n_in,
                              void* d_out, int out_size, void* d_ws, size_t ws_size,
                              hipStream_t stream) {
    const float* x    = (const float*)d_in[0];  // (B, S, I)
    const float* W    = (const float*)d_in[1];  // (I, 4H)
    const float* U    = (const float*)d_in[2];  // (H, 4H)
    const float* bias = (const float*)d_in[3];  // (4H,)
    float*       out  = (float*)d_out;          // (B, S, H)

    char*  ws  = (char*)d_ws;
    size_t off = 0;
    float*  xw  = (float*)(ws + off);  off += (size_t)S_ * B_ * G4_ * sizeof(float);
    __bf16* xb  = (__bf16*)(ws + off); off += (size_t)B_ * S_ * IP_ * sizeof(__bf16);
    __bf16* Wt  = (__bf16*)(ws + off); off += (size_t)G4_ * IP_ * sizeof(__bf16);
    __bf16* Ut  = (__bf16*)(ws + off); off += (size_t)G4_ * H_ * sizeof(__bf16);
    __bf16* h0  = (__bf16*)(ws + off); off += (size_t)B_ * H_ * sizeof(__bf16);
    __bf16* h1  = (__bf16*)(ws + off); off += (size_t)B_ * H_ * sizeof(__bf16);
    float*  cst = (float*)(ws + off);  off += (size_t)B_ * H_ * sizeof(float);

    cvt_x_bf16<<<((size_t)B_ * S_ * IP_ + 255) / 256, 256, 0, stream>>>(x, xb);
    cvt_w_bf16<<<((size_t)G4_ * IP_ + 255) / 256, 256, 0, stream>>>(W, Wt);
    cvt_u_bf16<<<((size_t)G4_ * H_ + 255) / 256, 256, 0, stream>>>(U, Ut);
    init_state<<<(B_ * H_ + 255) / 256, 256, 0, stream>>>(h0, h1, cst);

    lstm_xw_gemm<<<dim3((B_ * S_) / 32, G4_ / 512), 256, 0, stream>>>(xb, Wt, bias, xw);

    for (int s = 0; s < S_; ++s) {
        const __bf16* hin  = (s & 1) ? h1 : h0;
        __bf16*       hout = (s & 1) ? h0 : h1;
        lstm_step<<<dim3(B_ / 64, H_ / 16), 128, 0, stream>>>(xw, Ut, hin, hout, cst, out, s);
    }
}
